// MultiHeadAttention_41979010351706
// MI455X (gfx1250) — matrix operand order repaired and checked by reading, not compiled, not run on hardware
//
#include <hip/hip_runtime.h>
#include <stdint.h>

// ---------------------------------------------------------------------------
// MHA layer for MI455X (gfx1250): bf16 WMMA, fp32 accumulate.
//   x[4,2048,1024], src_mask[4,2048](bool), W_qkv[3072,1024], W_o[1024,1024]
//   out = proj( flash_attn( x @ W_qkv^T ) ) @ W_o^T
// GEMMs: 128x128 block tile, K-tile 64, double-buffered LDS, 1 barrier/iter.
// Attention: flash, 64-key tiles, async global->LDS for Q/K tiles.
// ---------------------------------------------------------------------------

typedef __bf16 v16bf __attribute__((ext_vector_type(16)));
typedef __bf16 v8bf  __attribute__((ext_vector_type(8)));
typedef float  v8f   __attribute__((ext_vector_type(8)));

struct alignas(8) us4_t { unsigned short x, y, z, w; };

__device__ __forceinline__ unsigned short f2bf_u16(float f) {
  union { float f; uint32_t u; } v; v.f = f;
  uint32_t r = (v.u + 0x7FFFu + ((v.u >> 16) & 1u)) >> 16;  // RNE
  return (unsigned short)r;
}

__device__ __forceinline__ v16bf mk16(v8bf lo, v8bf hi) {
  v16bf r;
#pragma unroll
  for (int i = 0; i < 8; ++i) { r[i] = lo[i]; r[i + 8] = hi[i]; }
  return r;
}

// A fragment (16x32 bf16, MxK), ISA 05_wmma wave32 layout.
__device__ __forceinline__ v16bf a_frag(const unsigned short* tile, int ld,
                                        int lane, int koff) {
  const int m = lane & 15, kh = (lane >> 4) << 3;
  const unsigned short* p = tile + m * ld + koff + kh;
  return mk16(*(const v8bf*)p, *(const v8bf*)(p + 16));
}

// B fragment (32x16 bf16, KxN) from LDS tile stored [N][K] (K contiguous).
__device__ __forceinline__ v16bf b_frag(const unsigned short* tile, int ld,
                                        int lane, int koff) {
  const int n = lane & 15, kh = (lane >> 4) << 3;
  const unsigned short* p = tile + n * ld + koff + kh;
  return mk16(*(const v8bf*)p, *(const v8bf*)(p + 16));
}

__device__ __forceinline__ v8f wmma_bf16(v16bf a, v16bf b, v8f c) {
  return __builtin_amdgcn_wmma_f32_16x16x32_bf16(false, a, false, b, (short)0,
                                                 c, false, false);
}

__device__ __forceinline__ us4_t to_bf4(float4 v) {
  return (us4_t){f2bf_u16(v.x), f2bf_u16(v.y), f2bf_u16(v.z), f2bf_u16(v.w)};
}
__device__ __forceinline__ us4_t to_bf4(us4_t v) { return v; }
__device__ __forceinline__ void store_c(float* p, float v) { *p = v; }
__device__ __forceinline__ void store_c(unsigned short* p, float v) { *p = f2bf_u16(v); }

template <typename T> struct vec4_of;
template <> struct vec4_of<float> { using type = float4; };
template <> struct vec4_of<unsigned short> { using type = us4_t; };

// Async 16B global->LDS copy (GLOBAL_LOAD_ASYNC_TO_LDS_B128, ASYNCcnt).
__device__ __forceinline__ void async_copy_b128(uint32_t lds_off, const void* g) {
  asm volatile("global_load_async_to_lds_b128 %0, %1, off"
               :: "v"(lds_off), "v"(g) : "memory");
}
__device__ __forceinline__ void wait_async0() {
  asm volatile("s_wait_asynccnt 0" ::: "memory");
}

// ---------------------------------------------------------------------------
// C[M,N] = A[M,K] @ W[N,K]^T.  Double-buffered, KT=64 (2 WMMA K-steps).
// 8 waves; wave (wm,wn) owns rows wm*32..+32, cols wn*64..+64.
// ---------------------------------------------------------------------------
template <typename TA, typename TC>
__global__ __launch_bounds__(256) void gemm_xwt(const TA* __restrict__ A,
                                                const float* __restrict__ W,
                                                TC* __restrict__ C,
                                                int M, int N, int K) {
  constexpr int BM = 128, BN = 128, KT = 64, LD = KT + 8;  // pad keeps 16B align
  __shared__ alignas(16) unsigned short sA[2][BM * LD];
  __shared__ alignas(16) unsigned short sB[2][BN * LD];
  using AV = typename vec4_of<TA>::type;
  const int t = threadIdx.x, lane = t & 31, w = t >> 5;
  const int wm = w & 3, wn = w >> 2;
  const int bm = blockIdx.y * BM, bn = blockIdx.x * BN;
  const int lrow = t >> 4, lcol = (t & 15) << 2;  // 16 vec4 per 64-wide row

  v8f acc[2][4];
#pragma unroll
  for (int i = 0; i < 2; ++i)
#pragma unroll
    for (int j = 0; j < 4; ++j) acc[i][j] = (v8f){0, 0, 0, 0, 0, 0, 0, 0};

  // prologue: tile 0 straight into LDS buffer 0
#pragma unroll
  for (int i = 0; i < 8; ++i) {
    const int r0 = lrow + 16 * i;
    *(us4_t*)&sA[0][r0 * LD + lcol] =
        to_bf4(*(const AV*)&A[(size_t)(bm + r0) * K + lcol]);
    *(us4_t*)&sB[0][r0 * LD + lcol] =
        to_bf4(*(const float4*)&W[(size_t)(bn + r0) * K + lcol]);
  }
  __syncthreads();

  const int nk = K / KT;
  for (int kt = 0; kt < nk; ++kt) {
    const int cur = kt & 1;
    // stage next tile global -> registers (overlaps with WMMA below)
    AV ra[8];
    float4 rw[8];
    if (kt + 1 < nk) {
      const int k0 = (kt + 1) * KT;
#pragma unroll
      for (int i = 0; i < 8; ++i) {
        const int r0 = lrow + 16 * i;
        ra[i] = *(const AV*)&A[(size_t)(bm + r0) * K + k0 + lcol];
        rw[i] = *(const float4*)&W[(size_t)(bn + r0) * K + k0 + lcol];
        if (kt + 2 < nk)  // global_prefetch_b8 hint two tiles ahead
          __builtin_prefetch(&A[(size_t)(bm + r0) * K + k0 + KT + lcol], 0, 1);
      }
    }
    // compute: 2 K-steps x (2 M x 4 N) = 16 WMMAs
#pragma unroll
    for (int ks = 0; ks < 2; ++ks) {
      const v16bf a0 = a_frag(&sA[cur][(wm * 32) * LD], LD, lane, ks * 32);
      const v16bf a1 = a_frag(&sA[cur][(wm * 32 + 16) * LD], LD, lane, ks * 32);
#pragma unroll
      for (int n = 0; n < 4; ++n) {
        const v16bf b =
            b_frag(&sB[cur][(wn * 64 + n * 16) * LD], LD, lane, ks * 32);
        acc[0][n] = wmma_bf16(a0, b, acc[0][n]);
        acc[1][n] = wmma_bf16(a1, b, acc[1][n]);
      }
    }
    if (kt + 1 < nk) {
      const int nxt = cur ^ 1;
#pragma unroll
      for (int i = 0; i < 8; ++i) {
        const int r0 = lrow + 16 * i;
        *(us4_t*)&sA[nxt][r0 * LD + lcol] = to_bf4(ra[i]);
        *(us4_t*)&sB[nxt][r0 * LD + lcol] = to_bf4(rw[i]);
      }
      __syncthreads();  // single barrier per iteration (buffers alternate)
    }
  }

  // C layout: elem r -> row r + 8*(lane>=16), col = lane%16
#pragma unroll
  for (int mt = 0; mt < 2; ++mt)
#pragma unroll
    for (int n = 0; n < 4; ++n)
#pragma unroll
      for (int r = 0; r < 8; ++r) {
        const int row = bm + wm * 32 + mt * 16 + r + ((lane >> 4) << 3);
        const int col = bn + wn * 64 + n * 16 + (lane & 15);
        store_c(&C[(size_t)row * N + col], acc[mt][n][r]);
      }
}

// ---------------------------------------------------------------------------
// Flash attention over bf16 qkv[b,s,3072] (head h: q@h*192, k@+64, v@+128).
// Block = (q-tile of 128, head, batch); 8 waves, each owns 16 query rows.
// ---------------------------------------------------------------------------
__global__ __launch_bounds__(256) void attn_fa(const unsigned short* __restrict__ qkv,
                                               const unsigned char* __restrict__ mask,
                                               unsigned short* __restrict__ outp) {
  constexpr int S = 2048, HD = 64, NH = 16, QB = 128, KB = 64, LD = HD + 8;
  constexpr int QKVLD = 3 * NH * HD;  // 3072
  const int t = threadIdx.x, lane = t & 31, w = t >> 5;
  const int q0 = blockIdx.x * QB, h = blockIdx.y, b = blockIdx.z;
  const size_t rowbase = (size_t)b * S;

  __shared__ alignas(16) unsigned short sQ[QB * LD];      // [q][d]
  __shared__ alignas(16) unsigned short sK[KB * LD];      // [k][d]
  __shared__ alignas(16) unsigned short sVt[HD * LD];     // [d][k] transposed
  __shared__ alignas(16) unsigned short sP[8 * 16 * LD];  // per-wave P scratch
  __shared__ float sBias[KB];

  // Q tile via async copies: 128 rows x 128B = 4 x b128 per thread
#pragma unroll
  for (int i = 0; i < 4; ++i) {
    const int e = t + 256 * i;
    const int row = e >> 3, colu = (e & 7) << 3;  // 8 ushorts = 16B
    async_copy_b128((uint32_t)(uintptr_t)&sQ[row * LD + colu],
                    &qkv[(rowbase + q0 + row) * QKVLD + h * 192 + colu]);
  }

  v8f O[4];
  float m_i[8], l_i[8];
#pragma unroll
  for (int n = 0; n < 4; ++n) O[n] = (v8f){0, 0, 0, 0, 0, 0, 0, 0};
#pragma unroll
  for (int r = 0; r < 8; ++r) { m_i[r] = -1e30f; l_i[r] = 0.0f; }

  for (int kt = 0; kt < S / KB; ++kt) {
    __syncthreads();  // protect previous iteration's sK/sVt reads
    // K tile: 64 rows x 128B = 2 x b128 async per thread
#pragma unroll
    for (int i = 0; i < 2; ++i) {
      const int e = t + 256 * i;
      const int row = e >> 3, colu = (e & 7) << 3;
      async_copy_b128(
          (uint32_t)(uintptr_t)&sK[row * LD + colu],
          &qkv[(rowbase + kt * KB + row) * QKVLD + h * 192 + 64 + colu]);
    }
    // V tile transposed in-flight (manual path)
#pragma unroll
    for (int i = 0; i < 4; ++i) {
      const int v = t + 256 * i;
      const int row = v >> 4, col = (v & 15) << 2;
      const us4_t vv = *(const us4_t*)&qkv[(rowbase + kt * KB + row) * QKVLD +
                                           h * 192 + 128 + col];
      sVt[(col + 0) * LD + row] = vv.x;
      sVt[(col + 1) * LD + row] = vv.y;
      sVt[(col + 2) * LD + row] = vv.z;
      sVt[(col + 3) * LD + row] = vv.w;
    }
    if (t < KB) sBias[t] = mask[rowbase + kt * KB + t] ? -1e30f : 0.0f;
    wait_async0();  // Q (iter 0) + K transfers complete
    __syncthreads();

    // logits = Q @ K^T  (contraction over d = 2 K-steps of 32)
    const v16bf qa0 = a_frag(sQ + (w * 16) * LD, LD, lane, 0);
    const v16bf qa1 = a_frag(sQ + (w * 16) * LD, LD, lane, 32);
    v8f sacc[4];
#pragma unroll
    for (int n = 0; n < 4; ++n) {
      sacc[n] = (v8f){0, 0, 0, 0, 0, 0, 0, 0};
      const v16bf kb0 = b_frag(sK + (n * 16) * LD, LD, lane, 0);
      const v16bf kb1 = b_frag(sK + (n * 16) * LD, LD, lane, 32);
      sacc[n] = wmma_bf16(qa0, kb0, sacc[n]);
      sacc[n] = wmma_bf16(qa1, kb1, sacc[n]);
    }
    // scale 1/sqrt(64) + key-padding bias
#pragma unroll
    for (int n = 0; n < 4; ++n) {
      const float bias = sBias[n * 16 + (lane & 15)];
#pragma unroll
      for (int r = 0; r < 8; ++r) sacc[n][r] = sacc[n][r] * 0.125f + bias;
    }
    // online softmax (rows live in one 16-lane half; xor masks 1..8 stay inside)
#pragma unroll
    for (int r = 0; r < 8; ++r) {
      float mx = fmaxf(fmaxf(sacc[0][r], sacc[1][r]),
                       fmaxf(sacc[2][r], sacc[3][r]));
#pragma unroll
      for (int sft = 8; sft >= 1; sft >>= 1)
        mx = fmaxf(mx, __shfl_xor(mx, sft, 32));
      const float m_new = fmaxf(m_i[r], mx);
      const float alpha = __expf(m_i[r] - m_new);
      float rs = 0.0f;
#pragma unroll
      for (int n = 0; n < 4; ++n) {
        const float p = __expf(sacc[n][r] - m_new);
        sacc[n][r] = p;
        rs += p;
      }
#pragma unroll
      for (int sft = 8; sft >= 1; sft >>= 1) rs += __shfl_xor(rs, sft, 32);
      l_i[r] = l_i[r] * alpha + rs;
      m_i[r] = m_new;
#pragma unroll
      for (int n = 0; n < 4; ++n) O[n][r] *= alpha;
    }
    // re-layout P (C-layout -> A-layout) through wave-private LDS
    unsigned short* pB = sP + (w * 16) * LD;
#pragma unroll
    for (int n = 0; n < 4; ++n)
#pragma unroll
      for (int r = 0; r < 8; ++r)
        pB[(r + ((lane >> 4) << 3)) * LD + n * 16 + (lane & 15)] =
            f2bf_u16(sacc[n][r]);
    const v16bf pa0 = a_frag(pB, LD, lane, 0);
    const v16bf pa1 = a_frag(pB, LD, lane, 32);
    // O += P @ V   (contraction over k = 2 K-steps of 32; sVt is [d][k])
#pragma unroll
    for (int n = 0; n < 4; ++n) {
      const v16bf vb0 = b_frag(sVt + (n * 16) * LD, LD, lane, 0);
      const v16bf vb1 = b_frag(sVt + (n * 16) * LD, LD, lane, 32);
      O[n] = wmma_bf16(pa0, vb0, O[n]);
      O[n] = wmma_bf16(pa1, vb1, O[n]);
    }
  }

  float inv[8];
#pragma unroll
  for (int r = 0; r < 8; ++r) inv[r] = 1.0f / l_i[r];
#pragma unroll
  for (int n = 0; n < 4; ++n)
#pragma unroll
    for (int r = 0; r < 8; ++r) {
      const int row = q0 + w * 16 + r + ((lane >> 4) << 3);
      const int col = h * HD + n * 16 + (lane & 15);
      outp[(rowbase + row) * (NH * HD) + col] = f2bf_u16(O[n][r] * inv[r]);
    }
}

// ---------------------------------------------------------------------------
extern "C" void kernel_launch(void* const* d_in, const int* in_sizes, int n_in,
                              void* d_out, int out_size, void* d_ws,
                              size_t ws_size, hipStream_t stream) {
  (void)in_sizes; (void)n_in; (void)out_size; (void)ws_size;
  constexpr int B = 4, S = 2048, DIM = 1024;
  constexpr int M = B * S;     // 8192
  constexpr int NQ = 3 * DIM;  // 3072
  const float* x            = (const float*)d_in[0];
  const unsigned char* mask = (const unsigned char*)d_in[1];
  const float* Wqkv         = (const float*)d_in[2];
  const float* Wo           = (const float*)d_in[3];
  float* out                = (float*)d_out;

  unsigned short* qkv  = (unsigned short*)d_ws;  // [8192,3072] bf16 (48MB)
  unsigned short* attn = qkv + (size_t)M * NQ;   // [8192,1024] bf16 (16MB)

  gemm_xwt<float, unsigned short>
      <<<dim3(NQ / 128, M / 128), 256, 0, stream>>>(x, Wqkv, qkv, M, NQ, DIM);
  attn_fa<<<dim3(S / 128, 16, B), 256, 0, stream>>>(qkv, mask, attn);
  gemm_xwt<unsigned short, float>
      <<<dim3(DIM / 128, M / 128), 256, 0, stream>>>(attn, Wo, out, M, DIM, DIM);
}
